// BiRNNBowman_64974265254389
// MI455X (gfx1250) — matrix operand order repaired and checked by reading, not compiled, not run on hardware
//
#include <hip/hip_runtime.h>
#include <hip/hip_bf16.h>
#include <stdint.h>

// ---------------------------------------------------------------------------
// BiRNN (Bowman SNLI) on MI455X / gfx1250.
// Recurrent GEMMs via v_wmma_f32_16x16x32_bf16 (f32 accum). Weights packed
// bf16-transposed once per launch (13.6MB -> resident in 192MB L2).
// ---------------------------------------------------------------------------

typedef __attribute__((ext_vector_type(16))) __bf16 v16bf;
typedef __attribute__((ext_vector_type(8)))  float  v8f;

#define B_   256
#define T_   256
#define D_   300
#define H_   512
#define G4   2048          // 4*H
#define KX   320           // D padded to multiple of 32
#define KTOT 832           // KX + H
#define KCH  (KTOT / 32)   // 26 k-chunks
#define FF_  1024

__device__ __forceinline__ unsigned short f2bf(float f) {
  unsigned u = __float_as_uint(f);
  u += 0x7FFFu + ((u >> 16) & 1u);     // round-to-nearest-even
  return (unsigned short)(u >> 16);
}
__device__ __forceinline__ float bf2f(unsigned short h) {
  return __uint_as_float(((unsigned)h) << 16);
}
__device__ __forceinline__ float sigm(float x) {
  return 1.0f / (1.0f + __expf(-x));
}

union Frag16 { uint4 q[2]; v16bf v; unsigned short u[16]; };

// ---------------------------------------------------------------------------
// Weight packing: WT[l][n][k] bf16, k<KX from Wx (zero-pad 300..319),
// k>=KX from Wh. Row-major in k so B-tile loads are contiguous.
// ---------------------------------------------------------------------------
struct WPtrs { const float* wx[4]; const float* wh[4]; };

__global__ void pack_lstm_w(WPtrs wp, unsigned short* wt) {
  long idx = (long)blockIdx.x * blockDim.x + threadIdx.x;
  const long total = 4L * G4 * KTOT;
  if (idx >= total) return;
  int k = (int)(idx % KTOT);
  int n = (int)((idx / KTOT) % G4);
  int l = (int)(idx / ((long)KTOT * G4));
  float v = 0.0f;
  if (k < KX) { if (k < D_) v = wp.wx[l][(long)k * G4 + n]; }
  else        { v = wp.wh[l][(long)(k - KX) * G4 + n]; }
  wt[idx] = f2bf(v);
}

// Generic f32 [K][N] -> bf16 [N][K] transpose for the MLP weights.
__global__ void transpose_cvt(const float* src, unsigned short* dst, int K, int N) {
  long idx = (long)blockIdx.x * blockDim.x + threadIdx.x;
  if (idx >= (long)K * N) return;
  int k = (int)(idx % K);
  int n = (int)(idx / K);
  dst[(long)n * K + k] = f2bf(src[(long)k * N + n]);
}

__global__ void init_state(float* cst, unsigned short* h0) {
  int idx = blockIdx.x * blockDim.x + threadIdx.x;
  if (idx < 4 * B_ * H_) { cst[idx] = 0.0f; h0[idx] = 0; }
}

// ---------------------------------------------------------------------------
// One recurrent step for all 4 LSTMs.
//   grid = (16 gate-col blocks, 4 row blocks, 4 lstms), block = 256 (8 waves)
// Workgroup owns 64 batch rows x 32 gate-cols across all 4 gates, so the
// full LSTM pointwise update happens in-block after the GEMM.
// ---------------------------------------------------------------------------
struct StepArgs {
  const float* prem; const float* hyp;
  const int* plen;   const int* hlen;
  const unsigned short* wt;        // [4][G4][KTOT] bf16
  const float* bias[4];            // [4H] each
  float* cstate;                   // [4][B][H] f32
  const unsigned short* hin;       // [4][B][H] bf16 (prev step)
  unsigned short* hout;            // [4][B][H] bf16 (this step)
  int step;
};

__global__ __launch_bounds__(256) void lstm_step(StepArgs a) {
  __shared__ __align__(16) float zbuf[64 * 128];          // 32 KB (z scratch)
  unsigned short* sA = (unsigned short*)zbuf;             // [64][32] bf16 (aliased)
  unsigned short* sB = sA + 64 * 32;                      // [128][32] bf16 (aliased)

  const int tid  = threadIdx.x;
  const int nblk = blockIdx.x;      // gate-col block (32 cols of H)
  const int mblk = blockIdx.y;      // 64-row batch block
  const int l    = blockIdx.z;      // lstm: 0=p_fw 1=p_bw 2=h_fw 3=h_bw
  const int t_eff = (l & 1) ? (T_ - 1 - a.step) : a.step;
  const float* X  = (l < 2) ? a.prem : a.hyp;
  const int* len  = (l < 2) ? a.plen : a.hlen;
  const unsigned short* WT  = a.wt  + (long)l * G4 * KTOT;
  const unsigned short* hin = a.hin + (long)l * B_ * H_;

  const int w    = tid >> 5;        // wave id (wave32)
  const int lane = tid & 31;
  const int mrow = w & 3;           // 16-row block within the 64 rows
  const int half = w >> 2;          // gates {2*half, 2*half+1}

  v8f acc[4] = {{}, {}, {}, {}};

  for (int kc = 0; kc < KCH; ++kc) {
    const int k0 = kc * 32;
    __syncthreads();
    // --- stage A tile [64 rows][32 k] into LDS --------------------------
    {
      int r = tid >> 2, q = tid & 3;
      int row_g = mblk * 64 + r;
      if (k0 < KX) {                       // from x_t (f32 -> bf16, pad >=300)
        const float* src = X + ((long)row_g * T_ + t_eff) * D_;
#pragma unroll
        for (int e = 0; e < 8; ++e) {
          int k = k0 + q * 8 + e;
          float v = (k < D_) ? src[k] : 0.0f;
          sA[r * 32 + q * 8 + e] = f2bf(v);
        }
      } else {                             // from h (already bf16)
        const unsigned short* src = hin + (long)row_g * H_ + (k0 - KX) + q * 8;
        *(uint4*)&sA[r * 32 + q * 8] = *(const uint4*)src;
      }
    }
    // --- stage B tile [4 gates * 32 n][32 k] into LDS -------------------
    {
      int rb = tid >> 1, hh = tid & 1;     // rb: 0..127
      int g = rb >> 5, j = rb & 31;
      long n_g = (long)g * H_ + nblk * 32 + j;
      const unsigned short* src = WT + n_g * KTOT + k0 + hh * 16;
      uint4* dst = (uint4*)&sB[rb * 32 + hh * 16];
      dst[0] = ((const uint4*)src)[0];
      dst[1] = ((const uint4*)src)[1];
    }
    __syncthreads();
    // --- A fragment (ISA 16-bit A 16x32 layout) -------------------------
    Frag16 af;
    {
      int row = mrow * 16 + (lane & 15);
      int ko  = (lane >> 4) * 8;
      af.q[0] = *(const uint4*)&sA[row * 32 + ko];
      af.q[1] = *(const uint4*)&sA[row * 32 + ko + 16];
    }
    // --- 4 WMMAs: gates {2*half,2*half+1} x 16-col subtiles {0,1} -------
#pragma unroll
    for (int p = 0; p < 4; ++p) {
      int g2   = half * 2 + (p >> 1);
      int nsub = p & 1;
      Frag16 bfr;
      int nrow = g2 * 32 + nsub * 16 + (lane & 15);
      int kb   = (lane >> 4) * 8;
      bfr.q[0] = *(const uint4*)&sB[nrow * 32 + kb];
      bfr.q[1] = *(const uint4*)&sB[nrow * 32 + kb + 16];
      acc[p] = __builtin_amdgcn_wmma_f32_16x16x32_bf16(
          false, af.v, false, bfr.v, (short)0, acc[p], false, false);
    }
  }
  __syncthreads();
  // --- dump accumulators to zbuf [64 rows][4*32 cols] (C layout §7.12.2) ---
#pragma unroll
  for (int p = 0; p < 4; ++p) {
    int g2   = half * 2 + (p >> 1);
    int nsub = p & 1;
#pragma unroll
    for (int r = 0; r < 8; ++r) {
      int row = mrow * 16 + (lane >> 4) * 8 + r;
      int col = g2 * 32 + nsub * 16 + (lane & 15);
      zbuf[row * 128 + col] = acc[p][r];
    }
  }
  __syncthreads();
  // --- LSTM pointwise update (i,j,f,o) with sequence-length masking --------
  const float* bias = a.bias[l];
  float* cst = a.cstate + (long)l * B_ * H_;
  unsigned short* hout = a.hout + (long)l * B_ * H_;
#pragma unroll
  for (int ii = 0; ii < 8; ++ii) {
    int idx = tid * 8 + ii;                // 64*32 = 2048 elements
    int row = idx >> 5, c = idx & 31;
    int row_g = mblk * 64 + row;
    int ncol  = nblk * 32 + c;
    float zi = zbuf[row * 128 + c]      + bias[ncol];
    float zj = zbuf[row * 128 + 32 + c] + bias[H_ + ncol];
    float zf = zbuf[row * 128 + 64 + c] + bias[2 * H_ + ncol];
    float zo = zbuf[row * 128 + 96 + c] + bias[3 * H_ + ncol];
    long sidx = (long)row_g * H_ + ncol;
    float cold = cst[sidx];
    float hold = bf2f(hin[sidx]);
    float nc = cold * sigm(zf + 1.0f) + sigm(zi) * tanhf(zj);
    float nh = tanhf(nc) * sigm(zo);
    bool m = (t_eff < len[row_g]);
    cst[sidx]  = m ? nc : cold;
    hout[sidx] = f2bf(m ? nh : hold);
  }
}

// ---------------------------------------------------------------------------
// Gather final cell states -> MLP input x = [c_pfw | c_pbw | c_hfw | c_hbw]
// ---------------------------------------------------------------------------
__global__ void pack_x(const float* cst, unsigned short* act) {
  int idx = blockIdx.x * blockDim.x + threadIdx.x;
  if (idx >= B_ * G4) return;
  int b = idx / G4, col = idx % G4;
  int l = col >> 9, n = col & 511;
  act[idx] = f2bf(cst[((long)l * B_ + b) * H_ + n]);
}

// ---------------------------------------------------------------------------
// MLP layer: out = tanh(A @ W + bias), A bf16 [256][K], WT bf16 [N][K].
//   grid = (N/128, 256/64), block = 256 (8 waves, each wave 16x64 of C)
// ---------------------------------------------------------------------------
__global__ __launch_bounds__(256) void mlp_gemm_tanh(
    const unsigned short* A, const unsigned short* WT, const float* bias,
    unsigned short* out, int K, int N) {
  __shared__ __align__(16) unsigned short sA[64 * 32];
  __shared__ __align__(16) unsigned short sB[128 * 32];
  const int tid  = threadIdx.x;
  const int nblk = blockIdx.x, mblk = blockIdx.y;
  const int w = tid >> 5, lane = tid & 31;
  const int mrow = w & 3, half = w >> 2;

  v8f acc[4] = {{}, {}, {}, {}};
  for (int k0 = 0; k0 < K; k0 += 32) {
    __syncthreads();
    { int r = tid >> 2, q = tid & 3;
      int row_g = mblk * 64 + r;
      *(uint4*)&sA[r * 32 + q * 8] = *(const uint4*)&A[(long)row_g * K + k0 + q * 8]; }
    { int rb = tid >> 1, hh = tid & 1;
      long n_g = (long)nblk * 128 + rb;
      const unsigned short* src = WT + n_g * K + k0 + hh * 16;
      uint4* d = (uint4*)&sB[rb * 32 + hh * 16];
      d[0] = ((const uint4*)src)[0];
      d[1] = ((const uint4*)src)[1]; }
    __syncthreads();
    Frag16 af;
    int row = mrow * 16 + (lane & 15);
    int ko  = (lane >> 4) * 8;
    af.q[0] = *(const uint4*)&sA[row * 32 + ko];
    af.q[1] = *(const uint4*)&sA[row * 32 + ko + 16];
#pragma unroll
    for (int p = 0; p < 4; ++p) {
      Frag16 bfr;
      int nrow = half * 64 + p * 16 + (lane & 15);
      int kb   = (lane >> 4) * 8;
      bfr.q[0] = *(const uint4*)&sB[nrow * 32 + kb];
      bfr.q[1] = *(const uint4*)&sB[nrow * 32 + kb + 16];
      acc[p] = __builtin_amdgcn_wmma_f32_16x16x32_bf16(
          false, af.v, false, bfr.v, (short)0, acc[p], false, false);
    }
  }
#pragma unroll
  for (int p = 0; p < 4; ++p) {
#pragma unroll
    for (int r = 0; r < 8; ++r) {
      int row_g = mblk * 64 + mrow * 16 + (lane >> 4) * 8 + r;
      int col   = nblk * 128 + half * 64 + p * 16 + (lane & 15);
      out[(long)row_g * N + col] = f2bf(tanhf(acc[p][r] + bias[col]));
    }
  }
}

// Final tiny projection [256,1024] @ [1024,3] + b4 -> f32 logits.
__global__ void final_logits(const unsigned short* h3, const float* W4,
                             const float* b4, float* out) {
  int b = blockIdx.x;
  int j = threadIdx.y;              // 0..2
  int lane = threadIdx.x;           // 0..31
  float s = 0.0f;
  for (int k = lane; k < FF_; k += 32)
    s += bf2f(h3[(long)b * FF_ + k]) * W4[(long)k * 3 + j];
#pragma unroll
  for (int off = 16; off > 0; off >>= 1) s += __shfl_down(s, off, 32);
  if (lane == 0) out[b * 3 + j] = s + b4[j];
}

// ---------------------------------------------------------------------------
extern "C" void kernel_launch(void* const* d_in, const int* in_sizes, int n_in,
                              void* d_out, int out_size, void* d_ws, size_t ws_size,
                              hipStream_t stream) {
  (void)in_sizes; (void)n_in; (void)out_size; (void)ws_size;
  const float* prem = (const float*)d_in[0];
  const float* hyp  = (const float*)d_in[1];
  const int* plen   = (const int*)d_in[2];
  const int* hlen   = (const int*)d_in[3];

  // Workspace layout (all sizes multiples of 256B).
  char* ws = (char*)d_ws;
  unsigned short* WT  = (unsigned short*)ws; ws += 4L * G4 * KTOT * 2;   // 13.6 MB
  unsigned short* W1T = (unsigned short*)ws; ws += (long)FF_ * G4 * 2;   // 4 MB
  unsigned short* W2T = (unsigned short*)ws; ws += (long)FF_ * FF_ * 2;  // 2 MB
  unsigned short* W3T = (unsigned short*)ws; ws += (long)FF_ * FF_ * 2;  // 2 MB
  float*          cst = (float*)ws;          ws += 4L * B_ * H_ * 4;     // 2 MB
  unsigned short* hbf = (unsigned short*)ws; ws += 2L * 4 * B_ * H_ * 2; // 2 MB (dbl buf)
  unsigned short* act0 = (unsigned short*)ws; ws += (long)B_ * G4 * 2;   // 1 MB
  unsigned short* act1 = (unsigned short*)ws; ws += (long)B_ * FF_ * 2;
  unsigned short* act2 = (unsigned short*)ws; ws += (long)B_ * FF_ * 2;
  unsigned short* act3 = (unsigned short*)ws; ws += (long)B_ * FF_ * 2;

  // --- pack weights (once per launch; deterministic) ----------------------
  WPtrs wp;
  wp.wx[0] = (const float*)d_in[4];  wp.wh[0] = (const float*)d_in[5];
  wp.wx[1] = (const float*)d_in[7];  wp.wh[1] = (const float*)d_in[8];
  wp.wx[2] = (const float*)d_in[10]; wp.wh[2] = (const float*)d_in[11];
  wp.wx[3] = (const float*)d_in[13]; wp.wh[3] = (const float*)d_in[14];
  {
    long tot = 4L * G4 * KTOT;
    pack_lstm_w<<<dim3((unsigned)((tot + 255) / 256)), 256, 0, stream>>>(wp, WT);
  }
  transpose_cvt<<<dim3((G4 * FF_ + 255) / 256), 256, 0, stream>>>(
      (const float*)d_in[16], W1T, G4, FF_);
  transpose_cvt<<<dim3((FF_ * FF_ + 255) / 256), 256, 0, stream>>>(
      (const float*)d_in[18], W2T, FF_, FF_);
  transpose_cvt<<<dim3((FF_ * FF_ + 255) / 256), 256, 0, stream>>>(
      (const float*)d_in[20], W3T, FF_, FF_);
  init_state<<<dim3((4 * B_ * H_ + 255) / 256), 256, 0, stream>>>(cst, hbf);

  // --- 256 recurrent steps, all 4 LSTMs fused per launch ------------------
  StepArgs sa;
  sa.prem = prem; sa.hyp = hyp; sa.plen = plen; sa.hlen = hlen;
  sa.wt = WT;
  sa.bias[0] = (const float*)d_in[6];
  sa.bias[1] = (const float*)d_in[9];
  sa.bias[2] = (const float*)d_in[12];
  sa.bias[3] = (const float*)d_in[15];
  sa.cstate = cst;
  const long HSZ = 4L * B_ * H_;
  for (int s = 0; s < T_; ++s) {
    sa.step = s;
    sa.hin  = hbf + (long)(s & 1) * HSZ;
    sa.hout = hbf + (long)((s + 1) & 1) * HSZ;
    lstm_step<<<dim3(16, 4, 4), 256, 0, stream>>>(sa);
  }

  // --- MLP head -----------------------------------------------------------
  pack_x<<<dim3((B_ * G4 + 255) / 256), 256, 0, stream>>>(cst, act0);
  mlp_gemm_tanh<<<dim3(FF_ / 128, B_ / 64), 256, 0, stream>>>(
      act0, W1T, (const float*)d_in[17], act1, G4, FF_);
  mlp_gemm_tanh<<<dim3(FF_ / 128, B_ / 64), 256, 0, stream>>>(
      act1, W2T, (const float*)d_in[19], act2, FF_, FF_);
  mlp_gemm_tanh<<<dim3(FF_ / 128, B_ / 64), 256, 0, stream>>>(
      act2, W3T, (const float*)d_in[21], act3, FF_, FF_);
  final_logits<<<dim3(B_), dim3(32, 3), 0, stream>>>(
      act3, (const float*)d_in[22], (const float*)d_in[23], (float*)d_out);
}
